// SE3Transformer_46368466927761
// MI455X (gfx1250) — matrix operand order repaired and checked by reading, not compiled, not run on hardware
//
#include <hip/hip_runtime.h>
#include <hip/hip_bf16.h>

// ---------------------------------------------------------------------------
// SE(3)-Transformer forward for MI455X (gfx1250, wave32, WMMA).
// Dense GEMM blocks use v_wmma_f32_16x16x32_bf16 (bf16 in, f32 accum).
// B operands are pre-swizzled to fragment order -> 2x global_load_b128/operand.
// A operands from row-major bf16 -> 2x b128 loads (two contiguous K-chunks).
// Scatter/softmax/norm paths stay fp32; feature tensor (41MB) is L2-resident.
// ---------------------------------------------------------------------------

typedef __attribute__((ext_vector_type(16))) __bf16 v16bf;
typedef __attribute__((ext_vector_type(8)))  float  v8f;

constexpr int kN = 20000;   // nodes
constexpr int kE = 320000;  // edges
constexpr int kC = 32;      // channels

__device__ __forceinline__ v8f wmma_bf16(v16bf a, v16bf b, v8f c) {
  return __builtin_amdgcn_wmma_f32_16x16x32_bf16(false, a, false, b, (short)0, c,
                                                 false, false);
}
__device__ __forceinline__ v8f zero8() {
  v8f z = {0.f, 0.f, 0.f, 0.f, 0.f, 0.f, 0.f, 0.f};
  return z;
}
// A-fragment element->K map (for gathered/strided sources)
__device__ __forceinline__ int a_kidx(int i, int lane) {
  return i + (i & 8) + ((lane & 16) >> 1);
}
// A fragment from a row-major bf16 row (>=32 elems): two contiguous 16B chunks
// elems 0..7 -> K = koff..koff+7 ; elems 8..15 -> K = 16+koff..23+koff
__device__ __forceinline__ v16bf load_A_rm(const __bf16* row) {
  int lane = threadIdx.x & 31;
  int koff = (lane & 16) >> 1;  // 0 or 8
  union { uint4 u[2]; v16bf v; } t;
  t.u[0] = *(const uint4*)(row + koff);
  t.u[1] = *(const uint4*)(row + 16 + koff);
  return t.v;
}
// B fragment from pre-swizzled weight: frag index selects (kTile,nTile);
// each lane reads its 16 contiguous bf16 (32 bytes).
__device__ __forceinline__ v16bf load_Bfrag(const __bf16* W, int frag) {
  int lane = threadIdx.x & 31;
  const uint4* p = (const uint4*)(W + ((size_t)frag * 32 + lane) * 16);
  union { uint4 u[2]; v16bf v; } t;
  t.u[0] = p[0];
  t.u[1] = p[1];
  return t.v;
}
// orderable-uint mapping for float atomic max
__device__ __forceinline__ unsigned fmap(float f) {
  unsigned b = __float_as_uint(f);
  return (b & 0x80000000u) ? ~b : (b | 0x80000000u);
}
__device__ __forceinline__ float funmap(unsigned u) {
  unsigned b = (u & 0x80000000u) ? (u ^ 0x80000000u) : ~u;
  return __uint_as_float(b);
}

// ---------------------------------------------------------------------------
// Utility kernels
// ---------------------------------------------------------------------------
__global__ void k_fill(float* p, float v, int n) {
  int i = blockIdx.x * blockDim.x + threadIdx.x;
  if (i < n) p[i] = v;
}

// f32 (Ks x Ns) -> bf16 B-fragment layout, zero padded to (Kd x Nd).
// dst linear index = ((kk*nT + t)*32 + lane)*16 + i ;
// source k = kk*32 + ((lane&16)>>1) + i + (i&8) , n = t*16 + (lane&15)
__global__ void k_cvt_frag(__bf16* dst, const float* src, int Kd, int Nd,
                           int Ks, int Ns) {
  int idx = blockIdx.x * blockDim.x + threadIdx.x;
  if (idx >= Kd * Nd) return;
  int i = idx & 15;
  int lane = (idx >> 4) & 31;
  int ft = idx >> 9;
  int nT = Nd >> 4;
  int kk = ft / nT, t = ft % nT;
  int k = kk * 32 + ((lane & 16) >> 1) + i + (i & 8);
  int n = t * 16 + (lane & 15);
  float v = (k < Ks && n < Ns) ? src[k * Ns + n] : 0.f;
  dst[idx] = (__bf16)v;
}

// feats layout: f[n*512 + c*16 + slot], slot base for degree l is l*l
__global__ void k_init_feats(float* f, const float* h) {
  int i = blockIdx.x * blockDim.x + threadIdx.x;
  if (i >= kN * 512) return;
  int n = i >> 9, r = i & 511, c = r >> 4, s = r & 15;
  f[i] = (s == 0) ? h[n * kC + c] : 0.f;
}

__global__ void k_extract_f0(__bf16* f0bf, const float* f) {
  int i = blockIdx.x * blockDim.x + threadIdx.x;
  if (i >= kN * kC) return;
  f0bf[i] = (__bf16)f[(size_t)(i >> 5) * 512 + (i & 31) * 16];
}

// ---------------------------------------------------------------------------
// Edge geometry: r, unit vec, 16 spherical harmonics, rad_in (K=32 bf16 pad)
// ---------------------------------------------------------------------------
__global__ void k_edge_geom(const float* x, const int* ei, const float* ea,
                            __bf16* radpad, float* yall) {
  int e = blockIdx.x * blockDim.x + threadIdx.x;
  if (e >= kE) return;
  int s = ei[e], d = ei[kE + e];
  float dx = x[s * 3 + 0] - x[d * 3 + 0];
  float dy = x[s * 3 + 1] - x[d * 3 + 1];
  float dz = x[s * 3 + 2] - x[d * 3 + 2];
  float r = sqrtf(dx * dx + dy * dy + dz * dz);
  float inv = 1.f / (r + 1e-8f);
  float ux = dx * inv, uy = dy * inv, uz = dz * inv;
  float xx = ux * ux, yy = uy * uy, zz = uz * uz;
  float xy = ux * uy, yz = uy * uz, xz = ux * uz;
  union { float f[16]; uint4 u[4]; } Y;
  Y.f[0] = 0.28209479177387814f;
  Y.f[1] = 0.4886025119029199f * uy;
  Y.f[2] = 0.4886025119029199f * uz;
  Y.f[3] = 0.4886025119029199f * ux;
  Y.f[4] = 1.0925484305920792f * xy;
  Y.f[5] = 1.0925484305920792f * yz;
  Y.f[6] = 0.31539156525252005f * (3.f * zz - 1.f);
  Y.f[7] = 1.0925484305920792f * xz;
  Y.f[8] = 0.5462742152960396f * (xx - yy);
  Y.f[9] = 0.5900435899266435f * uy * (3.f * xx - yy);
  Y.f[10] = 2.8906114426405543f * xy * uz;
  Y.f[11] = 0.45704579946446577f * uy * (5.f * zz - 1.f);
  Y.f[12] = 0.37317633259011546f * uz * (5.f * zz - 3.f);
  Y.f[13] = 0.45704579946446577f * ux * (5.f * zz - 1.f);
  Y.f[14] = 1.4453057213202771f * uz * (xx - yy);
  Y.f[15] = 0.5900435899266435f * ux * (xx - 3.f * yy);
  uint4* yd = (uint4*)(yall + (size_t)e * 16);
#pragma unroll
  for (int i = 0; i < 4; ++i) yd[i] = Y.u[i];
  union { __bf16 b[32]; uint4 u[4]; } rp;
  rp.b[0] = (__bf16)r;
  rp.b[1] = (__bf16)ea[e * 4 + 0];
  rp.b[2] = (__bf16)ea[e * 4 + 1];
  rp.b[3] = (__bf16)ea[e * 4 + 2];
  rp.b[4] = (__bf16)ea[e * 4 + 3];
#pragma unroll
  for (int i = 5; i < 32; ++i) rp.b[i] = (__bf16)0.f;
  uint4* rd = (uint4*)(radpad + (size_t)e * 32);
#pragma unroll
  for (int i = 0; i < 4; ++i) rd[i] = rp.u[i];
}

// ---------------------------------------------------------------------------
// qn = f0 @ Wq : one WMMA tile (16 nodes) per wave
// ---------------------------------------------------------------------------
__global__ __launch_bounds__(128) void k_qn(const __bf16* f0bf,
                                            const __bf16* Wq, float* qn) {
  int lane = threadIdx.x & 31;
  int tile = (blockIdx.x * 128 + threadIdx.x) >> 5;
  if (tile >= kN / 16) return;
  int base = tile * 16, m = lane & 15;
  v16bf a = load_A_rm(f0bf + (size_t)(base + m) * kC);
  v8f c0 = zero8(), c1 = zero8();
  c0 = wmma_bf16(a, load_Bfrag(Wq, 0), c0);
  c1 = wmma_bf16(a, load_Bfrag(Wq, 1), c1);
  int crow = (lane >> 4) << 3, ccol = lane & 15;
#pragma unroll
  for (int j = 0; j < 8; ++j) {
    qn[(base + crow + j) * kC + ccol] = c0[j];
    qn[(base + crow + j) * kC + 16 + ccol] = c1[j];
  }
}

// ---------------------------------------------------------------------------
// Per 16-edge tile: gather f0[src], K and V GEMMs (WMMA), store V (bf16),
// attention logits + segment atomic-max.
// ---------------------------------------------------------------------------
__global__ __launch_bounds__(128) void k_edge_kv(const __bf16* f0bf,
                                                 const int* ei,
                                                 const __bf16* Wk,
                                                 const __bf16* Wv,
                                                 const float* qn, __bf16* vbuf,
                                                 float* lbuf, unsigned* mu) {
  __shared__ float skt[4][16][32];
  int lane = threadIdx.x & 31, wv = threadIdx.x >> 5;
  int tile = blockIdx.x * 4 + wv, base = tile * 16;
  int m = lane & 15;
  int srow = ei[base + m];
  v16bf a = load_A_rm(f0bf + (size_t)srow * kC);
  v8f v0 = zero8(), v1 = zero8(), q0 = zero8(), q1 = zero8();
  v0 = wmma_bf16(a, load_Bfrag(Wv, 0), v0);
  v1 = wmma_bf16(a, load_Bfrag(Wv, 1), v1);
  q0 = wmma_bf16(a, load_Bfrag(Wk, 0), q0);
  q1 = wmma_bf16(a, load_Bfrag(Wk, 1), q1);
  int crow = (lane >> 4) << 3, ccol = lane & 15;
#pragma unroll
  for (int j = 0; j < 8; ++j) {
    int e = base + crow + j;
    vbuf[(size_t)e * kC + ccol] = (__bf16)v0[j];
    vbuf[(size_t)e * kC + 16 + ccol] = (__bf16)v1[j];
    skt[wv][crow + j][ccol] = q0[j];
    skt[wv][crow + j][16 + ccol] = q1[j];
  }
  __syncthreads();
  if (lane < 16) {
    int e = base + lane, d = ei[kE + e];
    float dot = 0.f;
#pragma unroll
    for (int c = 0; c < 32; ++c) dot += skt[wv][lane][c] * qn[d * kC + c];
    float lg = dot * 0.17677669529663687f;  // 1/sqrt(32)
    lbuf[e] = lg;
    atomicMax(&mu[d], fmap(lg));
  }
}

__global__ void k_softmax(float* lbuf, const unsigned* mu, float* sbuf,
                          const int* ei) {
  int e = blockIdx.x * blockDim.x + threadIdx.x;
  if (e >= kE) return;
  int d = ei[kE + e];
  float ev = __expf(lbuf[e] - funmap(mu[d]));
  lbuf[e] = ev;
  atomicAdd(&sbuf[d], ev);
}

// ---------------------------------------------------------------------------
// Self-interaction init: nxt[n,:,slot(l)] = cur[n,:,slot(l)] @ Ws[l]
// Rows are (node,slot) pairs grouped by degree l; 16-row WMMA tiles.
// ---------------------------------------------------------------------------
__global__ __launch_bounds__(128) void k_self_init(const float* cur,
                                                   const __bf16* Ws,
                                                   float* nxt, int lmax) {
  int lane = threadIdx.x & 31;
  int tile = (blockIdx.x * 128 + threadIdx.x) >> 5;
  int l, tbase;
  if (lmax == 1) {
    if (tile >= 1250) return;
    l = 0; tbase = 0;
  } else {
    if (tile < 1250)       { l = 0; tbase = 0; }
    else if (tile < 5000)  { l = 1; tbase = 1250; }
    else if (tile < 11250) { l = 2; tbase = 5000; }
    else if (tile < 20000) { l = 3; tbase = 11250; }
    else return;
  }
  int w21 = 2 * l + 1, sb = l * l;
  int rbase = (tile - tbase) * 16;
  int rid = rbase + (lane & 15);
  int n = rid / w21, slot = sb + rid % w21;
  v16bf a;
#pragma unroll
  for (int i = 0; i < 16; ++i)
    a[i] = (__bf16)cur[(size_t)n * 512 + a_kidx(i, lane) * 16 + slot];
  const __bf16* Wl = Ws + l * 1024;
  v8f c0 = zero8(), c1 = zero8();
  c0 = wmma_bf16(a, load_Bfrag(Wl, 0), c0);
  c1 = wmma_bf16(a, load_Bfrag(Wl, 1), c1);
  int crow = (lane >> 4) << 3, ccol = lane & 15;
#pragma unroll
  for (int j = 0; j < 8; ++j) {
    int r2 = rbase + crow + j;
    int n2 = r2 / w21, s2 = sb + r2 % w21;
    nxt[(size_t)n2 * 512 + ccol * 16 + s2] = c0[j];
    nxt[(size_t)n2 * 512 + (16 + ccol) * 16 + s2] = c1[j];
  }
}

// ---------------------------------------------------------------------------
// Fused radial-MLP (10 WMMAs/wave) + attention-weighted degree-wise scatter.
// One 16-edge tile per wave.
// ---------------------------------------------------------------------------
__global__ __launch_bounds__(128) void k_message(
    const __bf16* radpad, const __bf16* A, const float* br1, const float* br2,
    const __bf16* vbuf, const float* lbuf, const float* sbuf,
    const float* yall, const int* ei, float* nxt) {
  __shared__ __bf16 shid[4][16][32];
  __shared__ float sy[4][16][16];
  int lane = threadIdx.x & 31, wv = threadIdx.x >> 5;
  int tile = blockIdx.x * 4 + wv, base = tile * 16;
  __builtin_prefetch(radpad + (size_t)(base + 16) * 32, 0, 0);
  // stage Y tile (16 edges x 16 harmonics) with 2x b128 per lane
  {
    const uint4* ys = (const uint4*)(yall + (size_t)base * 16);
    uint4* yd = (uint4*)&sy[wv][0][0];
    yd[lane] = ys[lane];
    yd[lane + 32] = ys[lane + 32];
  }
  int m = lane & 15;
  v16bf ar = load_A_rm(radpad + (size_t)(base + m) * 32);
  v8f h0 = zero8(), h1 = zero8();
  h0 = wmma_bf16(ar, load_Bfrag(A, 0), h0);
  h1 = wmma_bf16(ar, load_Bfrag(A, 1), h1);
  int crow = (lane >> 4) << 3, ccol = lane & 15;
  float b0 = br1[ccol], b1 = br1[16 + ccol];
#pragma unroll
  for (int j = 0; j < 8; ++j) {
    shid[wv][crow + j][ccol] = (__bf16)fmaxf(h0[j] + b0, 0.f);
    shid[wv][crow + j][16 + ccol] = (__bf16)fmaxf(h1[j] + b1, 0.f);
  }
  __syncthreads();
  v16bf a2 = load_A_rm(&shid[wv][m][0]);
  const __bf16* Wr2 = A + 1024;  // 32x128 fragments
  int dstj[8];
  float alj[8], vv0[8], vv1[8];
#pragma unroll
  for (int j = 0; j < 8; ++j) {
    int e = base + crow + j;
    dstj[j] = ei[kE + e];
    alj[j] = lbuf[e] / (sbuf[dstj[j]] + 1e-8f);  // alpha
    vv0[j] = (float)vbuf[(size_t)e * kC + ccol];
    vv1[j] = (float)vbuf[(size_t)e * kC + 16 + ccol];
  }
#pragma unroll
  for (int t = 0; t < 8; ++t) {
    v8f w = zero8();
    w = wmma_bf16(a2, load_Bfrag(Wr2, t), w);
    const int l = t >> 1, sb = l * l, cnt = 2 * l + 1;
    int col = ccol + 16 * t, ch = col & 31;
    float b2 = br2[col];
#pragma unroll
    for (int j = 0; j < 8; ++j) {
      float wvv = (w[j] + b2) * alj[j] * ((t & 1) ? vv1[j] : vv0[j]);
      int row = crow + j;
      float* dp = nxt + (size_t)dstj[j] * 512 + ch * 16 + sb;
#pragma unroll
      for (int mm = 0; mm < cnt; ++mm)
        atomicAdd(dp + mm, wvv * sy[wv][row][sb + mm]);
    }
  }
}

// ---------------------------------------------------------------------------
// Gated norm: one wave per node, lane = channel; matvecs via lane shuffles.
// ---------------------------------------------------------------------------
__global__ __launch_bounds__(256) void k_gnorm(float* f, const float* Wn1,
                                               const float* Wn2,
                                               const float* bn1,
                                               const float* bn2) {
  int lane = threadIdx.x & 31;
  int node = (blockIdx.x * 256 + threadIdx.x) >> 5;
  if (node >= kN) return;
  float* fp = f + (size_t)node * 512 + lane * 16;
  float vals[16];
#pragma unroll
  for (int s = 0; s < 16; ++s) vals[s] = fp[s];
#pragma unroll
  for (int l = 0; l < 4; ++l) {
    int sb = l * l, w21 = 2 * l + 1;
    float ss = 0.f;
    for (int mm = 0; mm < w21; ++mm) ss += vals[sb + mm] * vals[sb + mm];
    float nrm = sqrtf(ss + 1e-8f);
    float s1 = bn1[l * 32 + lane];
    for (int c = 0; c < 32; ++c)
      s1 += __shfl(nrm, c, 32) * Wn1[l * 1024 + c * 32 + lane];
    s1 = fmaxf(s1, 0.f);
    float s2 = bn2[l * 32 + lane];
    for (int d = 0; d < 32; ++d)
      s2 += __shfl(s1, d, 32) * Wn2[l * 1024 + d * 32 + lane];
    float sc = s2 / nrm;
    for (int mm = 0; mm < w21; ++mm) vals[sb + mm] *= sc;
  }
#pragma unroll
  for (int s = 0; s < 16; ++s) fp[s] = vals[s];
}

// ---------------------------------------------------------------------------
// Final conv: radial MLP (32x144 padded), Wv0/Wv1 GEMMs, scatter agg0/agg1.
// ---------------------------------------------------------------------------
__global__ __launch_bounds__(128) void k_final_edge(
    const __bf16* radpad, const __bf16* FA, const float* br1f,
    const float* br2f, const __bf16* f0bf, const int* ei, const float* yall,
    float* agg0, float* agg1) {
  __shared__ __bf16 shid[4][16][32];
  int lane = threadIdx.x & 31, wv = threadIdx.x >> 5;
  int tile = blockIdx.x * 4 + wv, base = tile * 16;
  int m = lane & 15;
  int srow = ei[base + m];
  v16bf af = load_A_rm(f0bf + (size_t)srow * kC);
  v16bf ar = load_A_rm(radpad + (size_t)(base + m) * 32);
  v8f h0 = zero8(), h1 = zero8();
  h0 = wmma_bf16(ar, load_Bfrag(FA, 0), h0);
  h1 = wmma_bf16(ar, load_Bfrag(FA, 1), h1);
  int crow = (lane >> 4) << 3, ccol = lane & 15;
  float b0 = br1f[ccol], b1 = br1f[16 + ccol];
#pragma unroll
  for (int j = 0; j < 8; ++j) {
    shid[wv][crow + j][ccol] = (__bf16)fmaxf(h0[j] + b0, 0.f);
    shid[wv][crow + j][16 + ccol] = (__bf16)fmaxf(h1[j] + b1, 0.f);
  }
  __syncthreads();
  v16bf a2 = load_A_rm(&shid[wv][m][0]);
  const __bf16* Wr2p = FA + 1024;  // 32x144 fragments (130 padded)
  const __bf16* Wv0 = FA + 5632;   // 32x128 fragments
  const __bf16* Wv1p = FA + 9728;  // 32x16 fragments (2 padded)
  int dstj[8];
#pragma unroll
  for (int j = 0; j < 8; ++j) dstj[j] = ei[kE + base + crow + j];
  const float Y0c = 0.28209479177387814f;
#pragma unroll
  for (int t = 0; t < 8; ++t) {
    v8f wt = zero8(), vt = zero8();
    wt = wmma_bf16(a2, load_Bfrag(Wr2p, t), wt);
    vt = wmma_bf16(af, load_Bfrag(Wv0, t), vt);
    int col = ccol + 16 * t;
    float b2 = br2f[col];
#pragma unroll
    for (int j = 0; j < 8; ++j)
      atomicAdd(&agg0[(size_t)dstj[j] * 128 + col], (wt[j] + b2) * vt[j] * Y0c);
  }
  v8f w8 = zero8(), v1a = zero8();
  w8 = wmma_bf16(a2, load_Bfrag(Wr2p, 8), w8);
  v1a = wmma_bf16(af, load_Bfrag(Wv1p, 0), v1a);
  if (ccol < 2) {
    float b2 = br2f[128 + ccol];
#pragma unroll
    for (int j = 0; j < 8; ++j) {
      int e = base + crow + j;
      float wvv = (w8[j] + b2) * v1a[j];
#pragma unroll
      for (int mm = 0; mm < 3; ++mm)
        atomicAdd(&agg1[(size_t)dstj[j] * 6 + ccol * 3 + mm],
                  wvv * yall[(size_t)e * 16 + 1 + mm]);
    }
  }
}

// out0 = agg0 + f0@Wself0 ; h_pred = out0 @ Wmap + b   (chained WMMA GEMMs)
__global__ __launch_bounds__(32) void k_final_h(const __bf16* f0bf,
                                                const __bf16* FA,
                                                const float* agg0,
                                                const float* mapb, float* out) {
  __shared__ __bf16 so[16][128];
  int lane = threadIdx.x;
  int base = blockIdx.x * 16;
  int m = lane & 15;
  v16bf a = load_A_rm(f0bf + (size_t)(base + m) * kC);
  const __bf16* Wself0 = FA + 10240;  // 32x128 fragments
  int crow = (lane >> 4) << 3, ccol = lane & 15;
#pragma unroll
  for (int t = 0; t < 8; ++t) {
    v8f c = zero8();
    c = wmma_bf16(a, load_Bfrag(Wself0, t), c);
    int col = ccol + 16 * t;
#pragma unroll
    for (int j = 0; j < 8; ++j) {
      int row = crow + j;
      so[row][col] = (__bf16)(c[j] + agg0[(size_t)(base + row) * 128 + col]);
    }
  }
  __syncthreads();
  const __bf16* Wm = FA + 14336;  // 128x32 fragments: frag = kk*2 + t
  v8f h0 = zero8(), h1 = zero8();
#pragma unroll
  for (int kk = 0; kk < 4; ++kk) {
    v16bf a2 = load_A_rm(&so[m][kk * 32]);
    h0 = wmma_bf16(a2, load_Bfrag(Wm, kk * 2 + 0), h0);
    h1 = wmma_bf16(a2, load_Bfrag(Wm, kk * 2 + 1), h1);
  }
#pragma unroll
  for (int j = 0; j < 8; ++j) {
    int row = base + crow + j;
    out[row * 32 + ccol] = h0[j] + mapb[ccol];
    out[row * 32 + 16 + ccol] = h1[j] + mapb[16 + ccol];
  }
}

__global__ void k_final_x(const float* agg1, const float* cur,
                          const float* Wself1, const float* x, float* outx) {
  int i = blockIdx.x * blockDim.x + threadIdx.x;
  if (i >= kN * 3) return;
  int n = i / 3, mm = i % 3;
  float acc = agg1[(size_t)n * 6 + mm];  // d=0 row of out1
  const float* f1 = cur + (size_t)n * 512 + 1 + mm;  // degree-1 slots 1..3
  for (int c = 0; c < 32; ++c) acc += f1[c * 16] * Wself1[c * 2];
  outx[i] = acc + x[i];
}

// ---------------------------------------------------------------------------
// Host-side launch
// Input order: x, h, edge_index, edge_attr, then jax tree leaves of params
// (dict keys sorted): final{Wr1,Wr2,Wself0,Wself1,Wv0,Wv1,br1,br2},
// layers[0..3]{Wk,Wn1,Wn2,Wq,Wr1,Wr2,Ws,Wv,bn1,bn2,br1,br2}, map{W,b}.
// ---------------------------------------------------------------------------
extern "C" void kernel_launch(void* const* d_in, const int* in_sizes, int n_in,
                              void* d_out, int out_size, void* d_ws,
                              size_t ws_size, hipStream_t stream) {
  (void)in_sizes; (void)n_in; (void)out_size; (void)ws_size;
  const float* x = (const float*)d_in[0];
  const float* h = (const float*)d_in[1];
  const int* ei = (const int*)d_in[2];
  const float* ea = (const float*)d_in[3];
  const float* F_Wr1 = (const float*)d_in[4];
  const float* F_Wr2 = (const float*)d_in[5];
  const float* F_Wself0 = (const float*)d_in[6];
  const float* F_Wself1 = (const float*)d_in[7];
  const float* F_Wv0 = (const float*)d_in[8];
  const float* F_Wv1 = (const float*)d_in[9];
  const float* F_br1 = (const float*)d_in[10];
  const float* F_br2 = (const float*)d_in[11];
  const float *L_Wk[4], *L_Wn1[4], *L_Wn2[4], *L_Wq[4], *L_Wr1[4], *L_Wr2[4],
      *L_Ws[4], *L_Wv[4], *L_bn1[4], *L_bn2[4], *L_br1[4], *L_br2[4];
  for (int L = 0; L < 4; ++L) {
    int b = 12 + L * 12;
    L_Wk[L] = (const float*)d_in[b + 0];
    L_Wn1[L] = (const float*)d_in[b + 1];
    L_Wn2[L] = (const float*)d_in[b + 2];
    L_Wq[L] = (const float*)d_in[b + 3];
    L_Wr1[L] = (const float*)d_in[b + 4];
    L_Wr2[L] = (const float*)d_in[b + 5];
    L_Ws[L] = (const float*)d_in[b + 6];
    L_Wv[L] = (const float*)d_in[b + 7];
    L_bn1[L] = (const float*)d_in[b + 8];
    L_bn2[L] = (const float*)d_in[b + 9];
    L_br1[L] = (const float*)d_in[b + 10];
    L_br2[L] = (const float*)d_in[b + 11];
  }
  const float* mapW = (const float*)d_in[60];
  const float* mapb = (const float*)d_in[61];

  // ---- workspace carve ----
  char* p = (char*)d_ws;
  auto alloc = [&](size_t bytes) -> char* {
    char* r = p;
    p += (bytes + 255) & ~(size_t)255;
    return r;
  };
  float* yall = (float*)alloc((size_t)kE * 16 * 4);
  __bf16* radpad = (__bf16*)alloc((size_t)kE * 32 * 2);
  float* feats0 = (float*)alloc((size_t)kN * 512 * 4);
  float* feats1 = (float*)alloc((size_t)kN * 512 * 4);
  __bf16* vbuf = (__bf16*)alloc((size_t)kE * 32 * 2);
  float* lbuf = (float*)alloc((size_t)kE * 4);
  float* qn = (float*)alloc((size_t)kN * 32 * 4);
  unsigned* mu = (unsigned*)alloc((size_t)kN * 4);
  float* sbuf = (float*)alloc((size_t)kN * 4);
  __bf16* f0bf = (__bf16*)alloc((size_t)kN * 32 * 2);
  float* agg0 = (float*)alloc((size_t)kN * 128 * 4);
  float* agg1 = (float*)alloc((size_t)kN * 6 * 4);
  __bf16* arena = (__bf16*)alloc((size_t)67584 * 2);

  // ---- weight -> bf16 fragment-layout conversion (zero padded) ----
  auto cvt = [&](__bf16* dst, const float* src, int Kd, int Nd, int Ks, int Ns) {
    int tot = Kd * Nd;
    k_cvt_frag<<<(tot + 255) / 256, 256, 0, stream>>>(dst, src, Kd, Nd, Ks, Ns);
  };
  for (int L = 0; L < 4; ++L) {
    __bf16* A = arena + L * 12288;
    cvt(A + 0, L_Wr1[L], 32, 32, 5, 32);       // Wr1 K-padded to 32
    cvt(A + 1024, L_Wr2[L], 32, 128, 32, 128);
    cvt(A + 5120, L_Wq[L], 32, 32, 32, 32);
    cvt(A + 6144, L_Wk[L], 32, 32, 32, 32);
    cvt(A + 7168, L_Wv[L], 32, 32, 32, 32);
    for (int l = 0; l < 4; ++l)                // Ws: 4 independent 32x32
      cvt(A + 8192 + l * 1024, L_Ws[L] + l * 1024, 32, 32, 32, 32);
  }
  __bf16* FA = arena + 49152;
  cvt(FA + 0, F_Wr1, 32, 32, 5, 32);
  cvt(FA + 1024, F_Wr2, 32, 144, 32, 130);  // pad 130 -> 144 cols
  cvt(FA + 5632, F_Wv0, 32, 128, 32, 128);
  cvt(FA + 9728, F_Wv1, 32, 16, 32, 2);     // pad 2 -> 16 cols
  cvt(FA + 10240, F_Wself0, 32, 128, 32, 128);
  cvt(FA + 14336, mapW, 128, 32, 128, 32);  // 4 K-tiles x 2 N-tiles

  k_edge_geom<<<(kE + 255) / 256, 256, 0, stream>>>(x, ei, ea, radpad, yall);
  k_init_feats<<<(kN * 512 + 255) / 256, 256, 0, stream>>>(feats0, h);

  float* bufs[2] = {feats0, feats1};
  int ci = 0;
  for (int L = 0; L < 4; ++L) {
    float* cur = bufs[ci];
    float* nxt = bufs[ci ^ 1];
    __bf16* A = arena + L * 12288;
    k_fill<<<(kN * 512 + 255) / 256, 256, 0, stream>>>(nxt, 0.f, kN * 512);
    k_fill<<<(kN + 255) / 256, 256, 0, stream>>>((float*)mu, 0.f, kN);
    k_fill<<<(kN + 255) / 256, 256, 0, stream>>>(sbuf, 0.f, kN);
    k_extract_f0<<<(kN * 32 + 255) / 256, 256, 0, stream>>>(f0bf, cur);
    k_qn<<<(kN / 16 + 3) / 4, 128, 0, stream>>>(f0bf, A + 5120, qn);
    k_edge_kv<<<kE / 64, 128, 0, stream>>>(f0bf, ei, A + 6144, A + 7168, qn,
                                           vbuf, lbuf, mu);
    k_softmax<<<(kE + 255) / 256, 256, 0, stream>>>(lbuf, mu, sbuf, ei);
    int lmax = (L == 0) ? 1 : 4;
    int tiles = (L == 0) ? 1250 : 20000;
    k_self_init<<<(tiles + 3) / 4, 128, 0, stream>>>(cur, A + 8192, nxt, lmax);
    k_message<<<kE / 64, 128, 0, stream>>>(radpad, A, L_br1[L], L_br2[L], vbuf,
                                           lbuf, sbuf, yall, ei, nxt);
    k_gnorm<<<(kN * 32 + 255) / 256, 256, 0, stream>>>(nxt, L_Wn1[L], L_Wn2[L],
                                                       L_bn1[L], L_bn2[L]);
    ci ^= 1;
  }
  float* cur = bufs[ci];
  k_extract_f0<<<(kN * 32 + 255) / 256, 256, 0, stream>>>(f0bf, cur);
  k_fill<<<(kN * 128 + 255) / 256, 256, 0, stream>>>(agg0, 0.f, kN * 128);
  k_fill<<<(kN * 6 + 255) / 256, 256, 0, stream>>>(agg1, 0.f, kN * 6);
  k_final_edge<<<kE / 64, 128, 0, stream>>>(radpad, FA, F_br1, F_br2, f0bf, ei,
                                            yall, agg0, agg1);
  float* out = (float*)d_out;
  k_final_h<<<kN / 16, 32, 0, stream>>>(f0bf, FA, agg0, mapb, out);
  k_final_x<<<(kN * 3 + 255) / 256, 256, 0, stream>>>(agg1, cur, F_Wself1, x,
                                                      out + kN * 32);
}
